// SeqTagger_83734682402884
// MI455X (gfx1250) — matrix operand order repaired and checked by reading, not compiled, not run on hardware
//
#include <hip/hip_runtime.h>
#include <hip/hip_bf16.h>

// ---------------------------------------------------------------------------
// SeqTagger (biLSTM encoder + batch-as-sequence biLSTM decoder) for gfx1250.
//
// MI455X reasoning: encoder/decoder-pre GEMMs -> bf16 WMMA (fp32 accum).
// Encoder step kernel stages its shared A-tile in LDS via async-to-LDS
// (ASYNCcnt) so 4 waves sharing rows don't re-read global. Decoder's
// unavoidable serial matvec chain (2048x512 per step, weights resident in
// 192MB L2) uses V_PK_FMA_F32 to halve the per-step VALU chain length.
// ---------------------------------------------------------------------------

typedef unsigned short u16;
typedef unsigned int   u32;
typedef __attribute__((ext_vector_type(16))) __bf16 v16bf;
typedef __attribute__((ext_vector_type(8)))  __bf16 v8bf;
typedef __attribute__((ext_vector_type(8)))  float  v8f;
typedef __attribute__((ext_vector_type(2)))  float  v2f;
typedef int gv4i __attribute__((vector_size(16)));   // matches builtin param

#define DEV static __device__ __forceinline__
#define LDS_AS __attribute__((address_space(3)))

#if defined(__has_builtin)
#if __has_builtin(__builtin_amdgcn_global_load_async_to_lds_b128)
#define HAVE_ASYNC_LDS 1
#endif
#endif
#ifndef HAVE_ASYNC_LDS
#define HAVE_ASYNC_LDS 0
#endif

DEV float sigf(float x) { return 1.0f / (1.0f + __expf(-x)); }

DEV u16 f2bf(float f) {                       // round-to-nearest-even f32->bf16
  u32 u = __float_as_uint(f);
  u32 r = u + 0x7FFFu + ((u >> 16) & 1u);
  return (u16)(r >> 16);
}
DEV float bf2f(u16 h) { return __uint_as_float(((u32)h) << 16); }

// Stage one 16B chunk global -> LDS (async-to-LDS when available).
DEV void stage16(const u16* __restrict__ g, u16* l) {
#if HAVE_ASYNC_LDS
  __builtin_amdgcn_global_load_async_to_lds_b128((gv4i*)g, (LDS_AS gv4i*)l, 0,
                                                 0);
#else
  *(v8bf*)l = *(const v8bf*)g;
#endif
}

DEV void wait_async0() {
#if HAVE_ASYNC_LDS
#if __has_builtin(__builtin_amdgcn_s_wait_asynccnt)
  __builtin_amdgcn_s_wait_asynccnt(0);
#else
  asm volatile("s_wait_asynccnt 0" ::: "memory");
#endif
#endif
}

// A fragment, 16x32 bf16 (M x K), row-major source with leading dim `ld`.
// lane<16: row=lane,     elems 0..7 = K[k..k+7],    8..15 = K[k+16..k+23]
// lane>=16: row=lane-16, elems 0..7 = K[k+8..k+15], 8..15 = K[k+24..k+31]
DEV v16bf load_a(const u16* base, int ld, int row0, int k, int lane) {
  const u16* p = base + (size_t)(row0 + (lane & 15)) * ld + k + ((lane >> 4) << 3);
  v8bf lo = *(const v8bf*)p;
  v8bf hi = *(const v8bf*)(p + 16);
  v16bf r;
#pragma unroll
  for (int i = 0; i < 8; ++i) { r[i] = lo[i]; r[i + 8] = hi[i]; }
  return r;
}

// B fragment, 32x16 bf16 (K x N) where B[k][n] = W[n][k], W row-major (N x K,
// leading dim ldw) == PyTorch (4H, Din) weight layout. Two 16B loads per lane,
// same K order as load_a.
DEV v16bf load_b(const u16* __restrict__ w, int ldw, int n0, int k, int lane) {
  const u16* p = w + (size_t)(n0 + (lane & 15)) * ldw + k + ((lane >> 4) << 3);
  v8bf lo = *(const v8bf*)p;
  v8bf hi = *(const v8bf*)(p + 16);
  v16bf r;
#pragma unroll
  for (int i = 0; i < 8; ++i) { r[i] = lo[i]; r[i + 8] = hi[i]; }
  return r;
}

DEV v8f wmma_bf16(v16bf a, v16bf b, v8f c) {
  return __builtin_amdgcn_wmma_f32_16x16x32_bf16(false, a, false, b, (short)0, c,
                                                 false, false);
}

// ---------------------------------------------------------------------------
// Fused biLSTM encoder step: G = x_t @ Wih^T + h @ Whh^T + b, then gates.
// Block tile: 64 batch rows x 64 hidden cols. 8 waves; wave = 2 m-tiles x
// 4 gate-tiles (i,f,g,o). A k-tile (64x32 bf16, 4KB) staged in LDS via
// async-to-LDS; B fragments straight from global (L2-resident weights).
// ---------------------------------------------------------------------------
__global__ __launch_bounds__(256) void enc_step_kernel(
    const u16* __restrict__ xt, int din,            // (512, din) bf16
    const u16* __restrict__ wih,                    // (2048, din) bf16
    const u16* __restrict__ whh,                    // (2048, 512) bf16
    const float* __restrict__ bias,                 // (2048) fp32
    const u16* __restrict__ hprev,                  // (512, 512) bf16
    u16* __restrict__ hout,                         // (512, 512) bf16
    float* __restrict__ cbuf,                       // (512, 512) fp32, in/out
    u16* __restrict__ yout)                         // pre-offset, ld=1024
{
  __shared__ __align__(16) u16 Atile[64 * 32];      // one A k-tile
  const int tid  = threadIdx.x;
  const int lane = tid & 31;
  const int w    = tid >> 5;
  const int wm = w & 1, wc = w >> 1;
  const int Mblk = blockIdx.x * 64;                 // block's batch-row base
  const int M0   = Mblk + wm * 32;
  const int HB   = blockIdx.y * 64 + wc * 16;       // hidden cols

  const int srow = tid >> 2;                        // staging: 256 x 16B chunks
  const int scol = (tid & 3) * 8;
  u16* sdst = &Atile[srow * 32 + scol];

  v8f acc[2][4];
#pragma unroll
  for (int q = 0; q < 4; ++q) {
    float bv = bias[q * 512 + HB + (lane & 15)];
#pragma unroll
    for (int mt = 0; mt < 2; ++mt)
#pragma unroll
      for (int i = 0; i < 8; ++i) acc[mt][q][i] = bv;
  }

  for (int k = 0; k < din; k += 32) {               // input contribution
    stage16(xt + (size_t)(Mblk + srow) * din + k + scol, sdst);
    wait_async0();
    __syncthreads();
    v16bf a0 = load_a(Atile, 32, wm * 32, 0, lane);
    v16bf a1 = load_a(Atile, 32, wm * 32 + 16, 0, lane);
#pragma unroll
    for (int q = 0; q < 4; ++q) {
      v16bf b = load_b(wih, din, q * 512 + HB, k, lane);
      acc[0][q] = wmma_bf16(a0, b, acc[0][q]);
      acc[1][q] = wmma_bf16(a1, b, acc[1][q]);
    }
    __syncthreads();
  }
  for (int k = 0; k < 512; k += 32) {               // recurrent contribution
    stage16(hprev + (size_t)(Mblk + srow) * 512 + k + scol, sdst);
    wait_async0();
    __syncthreads();
    v16bf a0 = load_a(Atile, 32, wm * 32, 0, lane);
    v16bf a1 = load_a(Atile, 32, wm * 32 + 16, 0, lane);
#pragma unroll
    for (int q = 0; q < 4; ++q) {
      v16bf b = load_b(whh, 512, q * 512 + HB, k, lane);
      acc[0][q] = wmma_bf16(a0, b, acc[0][q]);
      acc[1][q] = wmma_bf16(a1, b, acc[1][q]);
    }
    __syncthreads();
  }

  const int hcol = HB + (lane & 15);                // C/D: N = lane%16
#pragma unroll
  for (int mt = 0; mt < 2; ++mt) {
#pragma unroll
    for (int r = 0; r < 8; ++r) {                   // C/D: M = r + 8*(lane>=16)
      int m = M0 + mt * 16 + r + ((lane >> 4) << 3);
      float iv = sigf(acc[mt][0][r]);
      float fv = sigf(acc[mt][1][r]);
      float gv = tanhf(acc[mt][2][r]);
      float ov = sigf(acc[mt][3][r]);
      float cn = fv * cbuf[m * 512 + hcol] + iv * gv;
      float hn = ov * tanhf(cn);
      cbuf[m * 512 + hcol] = cn;
      u16 hb = f2bf(hn);
      hout[m * 512 + hcol] = hb;
      yout[(size_t)m * 1024 + hcol] = hb;
    }
  }
}

// ---------------------------------------------------------------------------
// Dual-K-segment GEMM: out = A1 @ W1^T + A2 @ W2^T + bias (fp32 out).
// Block tile 128x128; wave = 2 m-tiles x 4 n-tiles.
// ---------------------------------------------------------------------------
__global__ __launch_bounds__(256) void gemm_dual_kernel(
    const u16* __restrict__ A1, int lda1, const u16* __restrict__ W1, int ldw1,
    int K1, const u16* __restrict__ A2, int lda2, const u16* __restrict__ W2,
    int ldw2, int K2, const float* __restrict__ bias, float* __restrict__ out,
    int ldo)
{
  const int lane = threadIdx.x & 31;
  const int w    = threadIdx.x >> 5;
  const int wm = w & 3, wn = w >> 2;
  const int M0 = blockIdx.x * 128 + wm * 32;
  const int N0 = blockIdx.y * 128 + wn * 64;

  v8f acc[2][4];
#pragma unroll
  for (int nt = 0; nt < 4; ++nt) {
    float bv = bias ? bias[N0 + nt * 16 + (lane & 15)] : 0.0f;
#pragma unroll
    for (int mt = 0; mt < 2; ++mt)
#pragma unroll
      for (int i = 0; i < 8; ++i) acc[mt][nt][i] = bv;
  }

  for (int k = 0; k < K1; k += 32) {
    if (k + 32 < K1)                                 // -> global_prefetch_b8
      __builtin_prefetch(A1 + (size_t)(M0 + (lane & 15)) * lda1 + k + 32, 0, 1);
    v16bf a0 = load_a(A1, lda1, M0, k, lane);
    v16bf a1 = load_a(A1, lda1, M0 + 16, k, lane);
#pragma unroll
    for (int nt = 0; nt < 4; ++nt) {
      v16bf b = load_b(W1, ldw1, N0 + nt * 16, k, lane);
      acc[0][nt] = wmma_bf16(a0, b, acc[0][nt]);
      acc[1][nt] = wmma_bf16(a1, b, acc[1][nt]);
    }
  }
  for (int k = 0; k < K2; k += 32) {
    v16bf a0 = load_a(A2, lda2, M0, k, lane);
    v16bf a1 = load_a(A2, lda2, M0 + 16, k, lane);
#pragma unroll
    for (int nt = 0; nt < 4; ++nt) {
      v16bf b = load_b(W2, ldw2, N0 + nt * 16, k, lane);
      acc[0][nt] = wmma_bf16(a0, b, acc[0][nt]);
      acc[1][nt] = wmma_bf16(a1, b, acc[1][nt]);
    }
  }

#pragma unroll
  for (int mt = 0; mt < 2; ++mt)
#pragma unroll
    for (int nt = 0; nt < 4; ++nt)
#pragma unroll
      for (int r = 0; r < 8; ++r) {
        int m = M0 + mt * 16 + r + ((lane >> 4) << 3);
        int n = N0 + nt * 16 + (lane & 15);
        out[(size_t)m * ldo + n] = acc[mt][nt][r];
      }
}

// ---------------------------------------------------------------------------
// Decoder serial scan over the batch dimension (1 block per direction).
// g[n] = pre[dir][b][n] + Whh[n,:] . h ; gates; state carried in LDS/regs.
// Inner dot uses V_PK_FMA_F32 (2 fp32 FMA/inst) to halve the serial chain.
// ---------------------------------------------------------------------------
__global__ __launch_bounds__(1024) void dec_scan_kernel(
    const float* __restrict__ pre,                  // (2, 512, 2048)
    const float* __restrict__ whh_f,                // (2048, 512) fp32
    const float* __restrict__ whh_b,
    float* __restrict__ hstate,                     // (2, 512)
    float* __restrict__ cstate,                     // (2, 512)
    u16* __restrict__ y1,                           // (512, 1024) bf16
    u16* __restrict__ y2)                           // optional copy (may be 0)
{
  __shared__ __align__(16) float hsh[512];
  __shared__ float gsh[2048];
  const int dir = blockIdx.x;
  const int tid = threadIdx.x;
  const float* whh = dir ? whh_b : whh_f;
  const float* prd = pre + (size_t)dir * 512 * 2048;
  float cj = 0.0f;
  if (tid < 512) { hsh[tid] = hstate[dir * 512 + tid]; cj = cstate[dir * 512 + tid]; }
  __syncthreads();

  for (int s = 0; s < 512; ++s) {
    int b = dir ? (511 - s) : s;
#pragma unroll
    for (int half = 0; half < 2; ++half) {
      int n = tid + half * 1024;
      const v2f* w2p = (const v2f*)(whh + (size_t)n * 512);
      const v2f* h2p = (const v2f*)hsh;
      v2f acc2 = {0.0f, 0.0f};
#pragma unroll 8
      for (int kk = 0; kk < 256; ++kk) {
        v2f wv = w2p[kk];                            // global_load_b64 (L2)
        v2f hv = h2p[kk];                            // ds_load_b64 (broadcast)
        asm("v_pk_fma_f32 %0, %1, %2, %0" : "+v"(acc2) : "v"(wv), "v"(hv));
      }
      gsh[n] = prd[(size_t)b * 2048 + n] + acc2.x + acc2.y;
    }
    __syncthreads();
    if (tid < 512) {
      float iv = sigf(gsh[tid]);
      float fv = sigf(gsh[512 + tid]);
      float gv = tanhf(gsh[1024 + tid]);
      float ov = sigf(gsh[1536 + tid]);
      cj = fv * cj + iv * gv;
      float hn = ov * tanhf(cj);
      hsh[tid] = hn;
      u16 hb = f2bf(hn);
      y1[(size_t)b * 1024 + dir * 512 + tid] = hb;
      if (y2) y2[(size_t)b * 1024 + dir * 512 + tid] = hb;
    }
    __syncthreads();
  }
  if (tid < 512) { hstate[dir * 512 + tid] = hsh[tid]; cstate[dir * 512 + tid] = cj; }
}

// --------------------------- small utility kernels -------------------------
__global__ void wconv_kernel(const float* __restrict__ src, int src_ld,
                             int src_off, u16* __restrict__ dst, int N, int Kin,
                             int Kpad) {
  int idx = blockIdx.x * blockDim.x + threadIdx.x;
  if (idx >= N * Kpad) return;
  int n = idx / Kpad, k = idx % Kpad;
  dst[idx] = (k < Kin) ? f2bf(src[(size_t)n * src_ld + src_off + k]) : (u16)0;
}

__global__ void embed_kernel(const int* __restrict__ tok,
                             const float* __restrict__ emb,
                             u16* __restrict__ xs) {
  int idx = blockIdx.x * blockDim.x + threadIdx.x;   // (T,B,320)
  if (idx >= 40 * 512 * 320) return;
  int e = idx % 320; int tb = idx / 320; int b = tb % 512; int t = tb / 512;
  int token = tok[b * 40 + t];                        // tokens are (B,T)
  xs[idx] = (e < 300) ? f2bf(emb[(size_t)token * 300 + e]) : (u16)0;
}

__global__ void zero_kernel(u32* __restrict__ p, int n) {
  int i = blockIdx.x * blockDim.x + threadIdx.x;
  if (i < n) p[i] = 0;
}

__global__ void fc_kernel(const u16* __restrict__ ys,      // (T,B,1024) bf16
                          const float* __restrict__ w,     // (10,1024)
                          const float* __restrict__ bias,  // (10)
                          float* __restrict__ out) {       // (B,T,10)
  int idx = blockIdx.x * blockDim.x + threadIdx.x;
  if (idx >= 512 * 40 * 10) return;
  int c = idx % 10; int bt = idx / 10; int t = bt % 40; int b = bt / 40;
  const u16* yv = ys + ((size_t)t * 512 + b) * 1024;
  const float* wr = w + c * 1024;
  float acc = bias[c];
  for (int k = 0; k < 1024; ++k) acc += bf2f(yv[k]) * wr[k];
  out[idx] = acc;
}

// ---------------------------------------------------------------------------
extern "C" void kernel_launch(void* const* d_in, const int* in_sizes, int n_in,
                              void* d_out, int out_size, void* d_ws,
                              size_t ws_size, hipStream_t stream) {
  (void)in_sizes; (void)n_in; (void)out_size; (void)ws_size;
  const int*   tok = (const int*)d_in[0];
  const float* emb = (const float*)d_in[1];
  const float *enc_wih[2][2], *enc_whh[2][2], *enc_b[2][2];
  const float *dec_wih[2][2], *dec_whh[2][2], *dec_b[2][2];
  int base = 2;
  for (int l = 0; l < 2; ++l)
    for (int d = 0; d < 2; ++d) {
      enc_wih[l][d] = (const float*)d_in[base++];
      enc_whh[l][d] = (const float*)d_in[base++];
      enc_b[l][d]   = (const float*)d_in[base++];
    }
  for (int l = 0; l < 2; ++l)
    for (int d = 0; d < 2; ++d) {
      dec_wih[l][d] = (const float*)d_in[base++];
      dec_whh[l][d] = (const float*)d_in[base++];
      dec_b[l][d]   = (const float*)d_in[base++];
    }
  const float* fcw = (const float*)d_in[26];
  const float* fcb = (const float*)d_in[27];

  // ---- workspace carving (bytes, 256B aligned) ----
  char* ws = (char*)d_ws;
  size_t off = 0;
  auto carve = [&](size_t bytes) -> char* {
    char* p = ws + off;
    off += (bytes + 255) & ~(size_t)255;
    return p;
  };
  u16 *e_wih[2][2], *e_whh[2][2], *d0_wa[2], *d0_wb[2], *d1_wi[2];
  for (int d = 0; d < 2; ++d) {
    e_wih[0][d] = (u16*)carve((size_t)2048 * 320 * 2);
    e_whh[0][d] = (u16*)carve((size_t)2048 * 512 * 2);
  }
  for (int d = 0; d < 2; ++d) {
    e_wih[1][d] = (u16*)carve((size_t)2048 * 1024 * 2);
    e_whh[1][d] = (u16*)carve((size_t)2048 * 512 * 2);
  }
  for (int d = 0; d < 2; ++d) {
    d0_wa[d] = (u16*)carve((size_t)2048 * 1024 * 2);   // Wih_d0 cols 0..1023
    d0_wb[d] = (u16*)carve((size_t)2048 * 1024 * 2);   // Wih_d0 cols 2048..3071
  }
  for (int d = 0; d < 2; ++d) d1_wi[d] = (u16*)carve((size_t)2048 * 1024 * 2);

  u16* xs0 = (u16*)carve((size_t)40 * 512 * 320 * 2);   // (T,B,320)
  u16* ys1 = (u16*)carve((size_t)40 * 512 * 1024 * 2);  // enc L0 out / dec out
  u16* ys2 = (u16*)carve((size_t)40 * 512 * 1024 * 2);  // enc L1 out
  u16* ysd = ys1;                                       // alias (L0 out consumed)
  u16* y0buf   = (u16*)carve((size_t)512 * 1024 * 2);
  u16* outprev = (u16*)carve((size_t)512 * 1024 * 2);
  u16* hA = (u16*)carve((size_t)512 * 512 * 2);
  u16* hB = (u16*)carve((size_t)512 * 512 * 2);
  float* cbuf = (float*)carve((size_t)512 * 512 * 4);
  float* pre0 = (float*)carve((size_t)2 * 512 * 2048 * 4);
  float* pre1 = (float*)carve((size_t)2 * 512 * 2048 * 4);
  float* decstate = (float*)carve((size_t)4096 * 4);
  float* dech0 = decstate, *decc0 = decstate + 1024;
  float* dech1 = decstate + 2048, *decc1 = decstate + 3072;

  auto zero = [&](void* p, int nu32) {
    zero_kernel<<<(nu32 + 255) / 256, 256, 0, stream>>>((u32*)p, nu32);
  };
  auto wc = [&](const float* src, int sld, int soff, u16* dst, int N, int Kin,
                int Kpad) {
    int n = N * Kpad;
    wconv_kernel<<<(n + 255) / 256, 256, 0, stream>>>(src, sld, soff, dst, N,
                                                      Kin, Kpad);
  };

  // ---- weight conversion fp32 -> bf16 (pad E 300->320; slice dec Wih) ----
  for (int d = 0; d < 2; ++d) {
    wc(enc_wih[0][d], 300, 0, e_wih[0][d], 2048, 300, 320);
    wc(enc_whh[0][d], 512, 0, e_whh[0][d], 2048, 512, 512);
    wc(enc_wih[1][d], 1024, 0, e_wih[1][d], 2048, 1024, 1024);
    wc(enc_whh[1][d], 512, 0, e_whh[1][d], 2048, 512, 512);
    wc(dec_wih[0][d], 3072, 0, d0_wa[d], 2048, 1024, 1024);
    wc(dec_wih[0][d], 3072, 2048, d0_wb[d], 2048, 1024, 1024);
    wc(dec_wih[1][d], 1024, 0, d1_wi[d], 2048, 1024, 1024);
  }

  // ---- embedding gather -> (T,B,320) bf16 ----
  {
    int n = 40 * 512 * 320;
    embed_kernel<<<(n + 255) / 256, 256, 0, stream>>>(tok, emb, xs0);
  }

  // ---- encoder: 2 layers x 2 directions x 40 fused WMMA+gate steps ----
  dim3 sgrid(8, 8), sblk(256);
  const u16* lin[2] = {xs0, ys1};
  const int  ldin[2] = {320, 1024};
  u16* lout[2] = {ys1, ys2};
  for (int l = 0; l < 2; ++l) {
    for (int d = 0; d < 2; ++d) {
      zero(hA, 512 * 512 / 2);           // bf16 -> u32 count
      zero(cbuf, 512 * 512);
      const u16* hp = hA;
      u16* hn = hB;
      for (int s = 0; s < 40; ++s) {
        int t = d ? (39 - s) : s;
        const u16* xt = lin[l] + (size_t)t * 512 * ldin[l];
        u16* yo = lout[l] + (size_t)t * 512 * 1024 + d * 512;
        enc_step_kernel<<<sgrid, sblk, 0, stream>>>(
            xt, ldin[l], e_wih[l][d], e_whh[l][d], enc_b[l][d], hp, hn, cbuf,
            yo);
        const u16* tmp = hp; hp = hn; hn = (u16*)tmp;
      }
    }
  }

  // ---- decoder: per outer t, batched WMMA pre-GEMMs + serial scans ----
  zero(outprev, 512 * 1024 / 2);
  zero(decstate, 4096);
  dim3 ggrid(4, 16), gblk(256);
  for (int t = 0; t < 40; ++t) {
    const u16* encx = ys2 + (size_t)t * 512 * 1024;
    for (int d = 0; d < 2; ++d)       // pre0 = encx@WihA^T + outprev@WihB^T + b
      gemm_dual_kernel<<<ggrid, gblk, 0, stream>>>(
          encx, 1024, d0_wa[d], 1024, 1024, outprev, 1024, d0_wb[d], 1024,
          1024, dec_b[0][d], pre0 + (size_t)d * 512 * 2048, 2048);
    dec_scan_kernel<<<2, 1024, 0, stream>>>(pre0, dec_whh[0][0], dec_whh[0][1],
                                            dech0, decc0, y0buf, (u16*)0);
    for (int d = 0; d < 2; ++d)       // pre1 = y0 @ Wih_d1^T + b
      gemm_dual_kernel<<<ggrid, gblk, 0, stream>>>(
          y0buf, 1024, d1_wi[d], 1024, 1024, y0buf, 1024, d1_wi[d], 1024, 0,
          dec_b[1][d], pre1 + (size_t)d * 512 * 2048, 2048);
    dec_scan_kernel<<<2, 1024, 0, stream>>>(pre1, dec_whh[1][0], dec_whh[1][1],
                                            dech1, decc1, outprev,
                                            ysd + (size_t)t * 512 * 1024);
  }

  // ---- final FC: (B,T,10) fp32 ----
  {
    int n = 512 * 40 * 10;
    fc_kernel<<<(n + 255) / 256, 256, 0, stream>>>(ysd, fcw, fcb,
                                                   (float*)d_out);
  }
}
